// HistogramLoss_9758165696887
// MI455X (gfx1250) — matrix operand order repaired and checked by reading, not compiled, not run on hardware
//
#include <hip/hip_runtime.h>
#include <hip/hip_bf16.h>

typedef __attribute__((ext_vector_type(16))) _Float16 v16h;
typedef __attribute__((ext_vector_type(8)))  float    v8f;

#define NBINS   64
#define QBINS   1280                 // fine bins over [-0.125, 1.125], width 1/1024
#define QSCALE  1024.0f
#define QOFF    0.125f
#define HW      65536                // 256*256
#define BATCH   8
#define NROWS   48                   // (b, image, channel) rows = 8*2*3
#define CNT_SCALE (1.0f / 1024.0f)   // keep counts exactly representable in f16

// ---------------------------------------------------------------------------
// Pass 1: masked fine-grained binning of both images (memory-bound pass).
// grid = 8 batches * 64 pixel chunks, 256 threads, 4 pixels/thread.
// LDS holds 6 (image,channel) fine histograms; flushed via global atomics.
// ---------------------------------------------------------------------------
__global__ __launch_bounds__(256)
void hist_fine_kernel(const float* __restrict__ gen,
                      const float* __restrict__ ref,
                      const float* __restrict__ mask,
                      unsigned int* __restrict__ counts,   // [NROWS][QBINS]
                      unsigned int* __restrict__ maskcnt)  // [BATCH]
{
    __shared__ unsigned int lds[6 * QBINS];
    __shared__ unsigned int msum;

    const int tid = threadIdx.x;
    for (int i = tid; i < 6 * QBINS; i += 256) lds[i] = 0u;
    if (tid == 0) msum = 0u;
    __syncthreads();

    const int b     = blockIdx.x >> 6;   // 64 chunks per batch sample
    const int chunk = blockIdx.x & 63;
    const int base  = chunk * 1024;

    unsigned int mlocal = 0;
    #pragma unroll
    for (int j = 0; j < 4; ++j) {
        const int n = base + tid + j * 256;
        const float m = mask[b * HW + n];
        if (m > 0.5f) {
            ++mlocal;
            #pragma unroll
            for (int c = 0; c < 3; ++c) {
                const int off = (b * 3 + c) * HW + n;
                float p = (gen[off] + 1.0f) * 0.5f;
                int   q = (int)fminf(fmaxf((p + QOFF) * QSCALE, 0.0f),
                                     (float)(QBINS - 1));
                atomicAdd(&lds[(0 * 3 + c) * QBINS + q], 1u);   // ds_add_u32
                p = (ref[off] + 1.0f) * 0.5f;
                q = (int)fminf(fmaxf((p + QOFF) * QSCALE, 0.0f),
                               (float)(QBINS - 1));
                atomicAdd(&lds[(1 * 3 + c) * QBINS + q], 1u);
            }
        }
    }
    if (mlocal) atomicAdd(&msum, mlocal);
    __syncthreads();

    // Flush LDS partials to global counts. Row index = 6*b + (image*3 + c).
    for (int idx = tid; idx < 6 * QBINS; idx += 256) {
        const unsigned int v = lds[idx];
        if (v) {
            const int ic  = idx / QBINS;
            const int q   = idx - ic * QBINS;
            const int row = b * 6 + ic;
            atomicAdd(&counts[row * QBINS + q], v);
        }
    }
    if (tid == 0 && msum) atomicAdd(&maskcnt[b], msum);
}

// ---------------------------------------------------------------------------
// Pass 2: hist[row, k] = sum_q counts[row, q] * G[q, k], G = Gaussian kernel,
// as a 48 x 1280 x 64 GEMM on the WMMA pipe, then normalize + L1 + reduce.
// One workgroup, 4 wave32s; wave w owns bin column tile [16w, 16w+16).
// ---------------------------------------------------------------------------
__global__ __launch_bounds__(128)
void hist_wmma_finalize_kernel(const unsigned int* __restrict__ counts,
                               const unsigned int* __restrict__ maskcnt,
                               float* __restrict__ out)
{
    __shared__ float histL[NROWS * NBINS];
    __shared__ float perb[BATCH];
    __shared__ float validL[BATCH];

    const int tid  = threadIdx.x;
    const int lane = tid & 31;
    const int wave = tid >> 5;        // column tile index (4 tiles of 16 bins)
    const int half = lane >> 4;       // 0: lanes 0-15, 1: lanes 16-31
    const int l16  = lane & 15;

    const int   binCol = wave * 16 + l16;
    const float centre = ((float)binCol + 0.5f) * (1.0f / (float)NBINS);

    v8f acc[3];
    #pragma unroll
    for (int rt = 0; rt < 3; ++rt) acc[rt] = v8f{0.f,0.f,0.f,0.f,0.f,0.f,0.f,0.f};

    for (int chunk = 0; chunk < QBINS / 32; ++chunk) {
        // B tile (32x16 f16): G[q, bin]; layout used here (contiguous k, same as A): lane n = lane&15,
        // lanes 0-15 hold K rows 0..15, lanes 16-31 hold K rows 16..31.
        v16h bm;
        #pragma unroll
        for (int e = 0; e < 16; ++e) {
            const int   q  = chunk * 32 + half * 16 + e;
            const float pq = ((float)q + 0.5f) * (1.0f / QSCALE) - QOFF;
            const float d  = (pq - centre) * (float)NBINS;   // distance in sigma
            bm[e] = (_Float16)__expf(-0.5f * d * d);
        }
        #pragma unroll
        for (int rt = 0; rt < 3; ++rt) {
            // A tile (16x32 f16): counts. Layout used here (contiguous k): row m = lane&15;
            // element e -> K = 16*half + e.
            v16h am;
            #pragma unroll
            for (int e = 0; e < 16; ++e) {
                const int k   = half * 16 + e;
                const int q   = chunk * 32 + k;
                const int row = rt * 16 + l16;
                am[e] = (_Float16)((float)counts[row * QBINS + q] * CNT_SCALE);
            }
            acc[rt] = __builtin_amdgcn_wmma_f32_16x16x32_f16(
                false, am, false, bm, (short)0, acc[rt], false, false);
        }
    }

    // D layout: VGPR r holds row (8*half + r), col = lane&15.
    #pragma unroll
    for (int rt = 0; rt < 3; ++rt) {
        #pragma unroll
        for (int r = 0; r < 8; ++r) {
            const int row = rt * 16 + half * 8 + r;
            histL[row * NBINS + binCol] = acc[rt][r];
        }
    }
    __syncthreads();

    // Per-(b,image,channel) normalization: hist / (sum_k + eps).
    if (tid < NROWS) {
        float s = 0.f;
        #pragma unroll 8
        for (int k = 0; k < NBINS; ++k) s += histL[tid * NBINS + k];
        const float inv = 1.0f / (s + 1e-8f);
        #pragma unroll 8
        for (int k = 0; k < NBINS; ++k) histL[tid * NBINS + k] *= inv;
    }
    __syncthreads();

    // Per-sample L1 over (c, k), masked validity, final mean.
    if (tid < BATCH) {
        float a = 0.f;
        for (int c = 0; c < 3; ++c) {
            const int rg = (tid * 2 + 0) * 3 + c;
            const int rr = (tid * 2 + 1) * 3 + c;
            #pragma unroll 8
            for (int k = 0; k < NBINS; ++k)
                a += fabsf(histL[rg * NBINS + k] - histL[rr * NBINS + k]);
        }
        const float valid = (maskcnt[tid] >= 10u) ? 1.0f : 0.0f;
        perb[tid]   = (a / (3.0f * (float)NBINS)) * valid;
        validL[tid] = valid;
    }
    __syncthreads();

    if (tid == 0) {
        float s = 0.f, nv = 0.f;
        #pragma unroll
        for (int i = 0; i < BATCH; ++i) { s += perb[i]; nv += validL[i]; }
        out[0] = (nv == 0.f) ? 0.f : s / fmaxf(nv, 1.0f);
    }
}

extern "C" void kernel_launch(void* const* d_in, const int* in_sizes, int n_in,
                              void* d_out, int out_size, void* d_ws, size_t ws_size,
                              hipStream_t stream) {
    (void)in_sizes; (void)n_in; (void)out_size;

    const float* gen  = (const float*)d_in[0];
    const float* ref  = (const float*)d_in[1];
    const float* mask = (const float*)d_in[2];

    unsigned int* counts  = (unsigned int*)d_ws;
    unsigned int* maskcnt = counts + NROWS * QBINS;
    const size_t ws_need = (size_t)(NROWS * QBINS + BATCH) * sizeof(unsigned int);
    if (ws_size < ws_need) return;  // ~246 KB scratch required

    hipMemsetAsync(d_ws, 0, ws_need, stream);

    hist_fine_kernel<<<dim3(BATCH * 64), dim3(256), 0, stream>>>(
        gen, ref, mask, counts, maskcnt);

    hist_wmma_finalize_kernel<<<dim3(1), dim3(128), 0, stream>>>(
        counts, maskcnt, (float*)d_out);
}
